// LSTMNetwork_25185688224320
// MI455X (gfx1250) — matrix operand order repaired and checked by reading, not compiled, not run on hardware
//
#include <hip/hip_runtime.h>

typedef __attribute__((ext_vector_type(16))) _Float16 v16h;
typedef __attribute__((ext_vector_type(8)))  _Float16 v8h;
typedef __attribute__((ext_vector_type(8)))  float    v8f;

constexpr int kHid     = 64;
constexpr int kBT      = 16;    // batch rows per workgroup (one WMMA M-tile)
constexpr int kT       = 512;
constexpr int kB       = 1024;
constexpr int kThreads = 128;   // 4 waves (wave32): wave w owns h-cols [16w,16w+16)

__device__ __forceinline__ v8f wmma_f16(v16h a, v16h b, v8f c) {
#if defined(__HIP_DEVICE_COMPILE__)
  // D = A(16x32 f16) * B(32x16 f16) + C(16x16 f32)
  return __builtin_amdgcn_wmma_f32_16x16x32_f16(false, a, false, b, (short)0, c,
                                                false, false);
#else
  (void)a; (void)b;
  return c;
#endif
}

// ---- branch-free activations: 1 TRANS op each via v_tanh_f32 ----
__device__ __forceinline__ float tanh_fast(float x) {
#if defined(__HIP_DEVICE_COMPILE__) && __has_builtin(__builtin_amdgcn_tanhf)
  return __builtin_amdgcn_tanhf(x);           // v_tanh_f32
#elif defined(__HIP_DEVICE_COMPILE__) && __has_builtin(__builtin_amdgcn_exp2f) && \
      __has_builtin(__builtin_amdgcn_rcpf)
  return 2.0f * __builtin_amdgcn_rcpf(
             1.0f + __builtin_amdgcn_exp2f(-2.8853900817779268f * x)) - 1.0f;
#else
  return tanhf(x);
#endif
}
// sigmoid(x + b) with pre-halved bias hb = b/2:
//   = 0.5*tanh(0.5*x + hb) + 0.5   (bias folds into the existing multiply -> FMA)
__device__ __forceinline__ float sigmoid_b(float x, float hb) {
  return __builtin_fmaf(0.5f, tanh_fast(__builtin_fmaf(0.5f, x, hb)), 0.5f);
}

// One LSTM layer, gate-permuted WMMA packing:
//   wave w owns h-cols j in [16w,16w+16); its 4 N-tiles are gate blocks
//   i/f/g/o for those SAME columns (global gate col n = 64*g + 16w + l15).
// => each lane's 4 accumulators hold i,f,g,o for its own (m,j) cells: the
//    LSTM cell update runs entirely in registers (no gate LDS round-trip).
// First WMMA of each chain uses inline-0 C; bias is folded into activations.
// h and x are ping-pong double-buffered in LDS -> ONE barrier per timestep.
// Timestep loop is kept at unroll 1 so the ~128 VGPRs of packed weight
// fragments stay resident (unroll-2 pipelining made the allocator remat them).
template <int IN_DIM, int KX_TILES, bool WRITE_SEQ, bool FUSE_FC>
__global__ __launch_bounds__(kThreads)
void lstm_layer(const void* __restrict__ xin,      // L0: f32 (B,T,IN); L1: f16 (T,B,64)
                const float* __restrict__ W_ih,    // (256, IN_DIM)
                const float* __restrict__ W_hh,    // (256, 64)
                const float* __restrict__ b_ih,    // (256)
                const float* __restrict__ b_hh,    // (256)
                _Float16* __restrict__ hseq,       // (T,B,64) f16 out (if WRITE_SEQ)
                const float* __restrict__ W_fc1,   // (32,64)  (if FUSE_FC)
                const float* __restrict__ b_fc1,   // (32)
                const float* __restrict__ W_fc2,   // (1,32)
                const float* __restrict__ b_fc2,   // (1)
                float* __restrict__ out) {         // (B,)     (if FUSE_FC)
  constexpr int XS = KX_TILES * 32 + 8;  // x_buf row stride (f16): conflict-free b128
  constexpr int HS = kHid + 8;           // 72 f16 = 144B rows -> distinct bank groups

  __shared__ __align__(16) _Float16 h_buf[2][kBT][HS];
  __shared__ __align__(16) _Float16 x_buf[2][kBT][XS];

  const int tid  = threadIdx.x;
  const int wave = tid >> 5;
  const int lane = tid & 31;
  const int hi   = (lane >> 4) & 1;
  const int l15  = lane & 15;
  const int b0   = blockIdx.x * kBT;
  const int jcol = wave * 16 + l15;      // h column owned by this lane

  // ---- loop-invariant permuted weight B-fragments + folded bias in VGPRs ----
  // B layout (32x16 f16): element e of v16h <-> K = kt*32 + 8*hi + (e&7) + 16*(e>>3), N = l15.
  v16h Bhh[2][4];            // [ktile][gate]
  v16h Bih[KX_TILES][4];
  float hbias[4];            // gates i,f,o: bias/2 (for sigmoid FMA); gate 2: full
  for (int g = 0; g < 4; ++g) {
    const int n = g * 64 + jcol;         // permuted gate column
    const float bv = b_ih[n] + b_hh[n];
    hbias[g] = (g == 2) ? bv : 0.5f * bv;
    for (int kt = 0; kt < 2; ++kt) {
      v16h f;
      for (int e = 0; e < 16; ++e)
        f[e] = (_Float16)W_hh[n * kHid + kt * 32 + (e & 7) + hi * 8 + (e >> 3) * 16];
      Bhh[kt][g] = f;
    }
    for (int kt = 0; kt < KX_TILES; ++kt) {
      v16h f;
      for (int e = 0; e < 16; ++e) {
        const int k = kt * 32 + (e & 7) + hi * 8 + (e >> 3) * 16;
        f[e] = (k < IN_DIM) ? (_Float16)W_ih[n * IN_DIM + k] : (_Float16)0.f;
      }
      Bih[kt][g] = f;
    }
  }

  // zero both h ping-pong buffers and x padding
  for (int i = tid; i < 2 * kBT * HS; i += kThreads) (&h_buf[0][0][0])[i] = (_Float16)0.f;
  for (int i = tid; i < 2 * kBT * XS; i += kThreads) (&x_buf[0][0][0])[i] = (_Float16)0.f;

  // coalesced copy indices (hseq output + layer1 staging): 16B per thread
  const int csm = tid >> 3, csd = (tid & 7) * 8;
  _Float16* hseqp = WRITE_SEQ ? &hseq[(size_t)(b0 + csm) * kHid + csd] : nullptr;

  // ---- x staging: register double buffer, committed into ping-pong LDS ----
  const bool stager = (IN_DIM == 6) ? (tid < kBT * 6) : true;
  int sm = 0, sd = 0;
  const float*    x0p = nullptr;
  const _Float16* x1p = nullptr;
  float x0r = 0.f;
  v8h   x1r = {};
  if (IN_DIM == 6) {
    sm = tid / 6; sd = tid - sm * 6;
    if (stager) {
      x0p = (const float*)xin + ((size_t)(b0 + sm) * kT) * 6 + sd;
      x_buf[0][sm][sd] = (_Float16)x0p[0];      // x_0
      x0r = x0p[6];                             // x_1
    }
  } else {
    x1p = (const _Float16*)xin + (size_t)(b0 + csm) * kHid + csd;
    *(v8h*)&x_buf[0][csm][csd] = *(const v8h*)x1p;                 // x_0
    x1r = *(const v8h*)(x1p + (size_t)kB * kHid);                  // x_1
  }

  // per-thread cell state: rows m = hi*8 + v at column jcol
  float c[8];
#pragma unroll
  for (int v = 0; v < 8; ++v) c[v] = 0.f;

  __syncthreads();

#pragma unroll 1
  for (int t = 0; t < kT; ++t) {
    const int p = t & 1;   // h_buf[p] = h_{t-1}, x_buf[p] = x_t (wave-uniform)

    // ---- coalesced hseq output of h_{t-1} (stable buffer, off critical path) --
    if (WRITE_SEQ && t > 0)
      *(v8h*)&hseqp[(size_t)(t - 1) * kB * kHid] = *(const v8h*)&h_buf[p][csm][csd];

    // ---- commit x_{t+1} into the other buffer; issue load of x_{t+2} ----
    if (IN_DIM == 6) {
      if (stager) {
        if (t + 1 < kT) x_buf[p ^ 1][sm][sd] = (_Float16)x0r;
        if (t + 2 < kT) x0r = x0p[(size_t)(t + 2) * 6];
      }
    } else {
      if (t + 1 < kT) *(v8h*)&x_buf[p ^ 1][csm][csd] = x1r;
      if (t + 2 < kT) x1r = *(const v8h*)(x1p + (size_t)(t + 2) * kB * kHid);
    }

    // ---- A fragments (shared across all 4 gate tiles of this wave) ----
    // A layout (16x32 f16): lane row M=l15; elems 0..7 K=kt*32+8*hi+(0..7),
    // elems 8..15 K=kt*32+16+8*hi+(0..7)  -> two contiguous b128 runs.
    // Ax loads issued FIRST: they feed the first WMMAs, enabling a partial
    // s_wait_dscnt while the Ah loads still fly.
    v16h Ah[2], Ax[KX_TILES];
#pragma unroll
    for (int kt = 0; kt < KX_TILES; ++kt) {
      const int kb = kt * 32 + hi * 8;
      v8h lo = *(const v8h*)&x_buf[p][l15][kb];
      v8h hh = *(const v8h*)&x_buf[p][l15][kb + 16];
      Ax[kt] = __builtin_shufflevector(lo, hh, 0,1,2,3,4,5,6,7,8,9,10,11,12,13,14,15);
    }
#pragma unroll
    for (int kt = 0; kt < 2; ++kt) {
      const int kb = kt * 32 + hi * 8;
      v8h lo = *(const v8h*)&h_buf[p][l15][kb];
      v8h hh = *(const v8h*)&h_buf[p][l15][kb + 16];
      Ah[kt] = __builtin_shufflevector(lo, hh, 0,1,2,3,4,5,6,7,8,9,10,11,12,13,14,15);
    }

    // ---- gates: 4 independent accumulator chains (i,f,g,o) ----
    // first WMMA per chain uses an all-zero C -> inline SRC2=0, no init movs
    v8f acc[4];
    const v8f zc = {};
#pragma unroll
    for (int g = 0; g < 4; ++g) acc[g] = wmma_f16(Ax[0], Bih[0][g], zc);
#pragma unroll
    for (int kt = 1; kt < KX_TILES; ++kt)
#pragma unroll
      for (int g = 0; g < 4; ++g) acc[g] = wmma_f16(Ax[kt], Bih[kt][g], acc[g]);
#pragma unroll
    for (int kt = 0; kt < 2; ++kt)
#pragma unroll
      for (int g = 0; g < 4; ++g) acc[g] = wmma_f16(Ah[kt], Bhh[kt][g], acc[g]);

    // ---- LSTM cell update in registers (C/D: M = v + 8*hi, N = l15) ----
    // bias folded: sigmoid via FMA with pre-halved bias, tanh with one add
#pragma unroll
    for (int v = 0; v < 8; ++v) {
      const float iv = sigmoid_b(acc[0][v], hbias[0]);
      const float fv = sigmoid_b(acc[1][v], hbias[1]);
      const float gv = tanh_fast(acc[2][v] + hbias[2]);
      const float ov = sigmoid_b(acc[3][v], hbias[3]);
      const float cn = fv * c[v] + iv * gv;
      c[v] = cn;
      h_buf[p ^ 1][hi * 8 + v][jcol] = (_Float16)(ov * tanh_fast(cn));  // b16 store
    }
    __syncthreads();   // single barrier per timestep
  }

  if (WRITE_SEQ)   // final timestep output (h_{T-1} lives in h_buf[kT&1])
    *(v8h*)&hseqp[(size_t)(kT - 1) * kB * kHid] = *(const v8h*)&h_buf[kT & 1][csm][csd];

  if (FUSE_FC) {
    // h_buf[kT&1] holds h_{T-1}; relu(h @ W_fc1^T + b_fc1) @ W_fc2^T + b_fc2
    if (tid < kBT) {
      float o = b_fc2[0];
      for (int f = 0; f < 32; ++f) {
        float a = b_fc1[f];
        for (int k = 0; k < kHid; ++k)
          a += (float)h_buf[kT & 1][tid][k] * W_fc1[f * kHid + k];
        o += fmaxf(a, 0.f) * W_fc2[f];
      }
      out[b0 + tid] = o;
    }
  }
}

extern "C" void kernel_launch(void* const* d_in, const int* in_sizes, int n_in,
                              void* d_out, int out_size, void* d_ws, size_t ws_size,
                              hipStream_t stream) {
  (void)in_sizes; (void)n_in; (void)out_size; (void)ws_size;
  const float* x     = (const float*)d_in[0];
  const float* W_ih0 = (const float*)d_in[1];
  const float* W_hh0 = (const float*)d_in[2];
  const float* b_ih0 = (const float*)d_in[3];
  const float* b_hh0 = (const float*)d_in[4];
  const float* W_ih1 = (const float*)d_in[5];
  const float* W_hh1 = (const float*)d_in[6];
  const float* b_ih1 = (const float*)d_in[7];
  const float* b_hh1 = (const float*)d_in[8];
  const float* W_fc1 = (const float*)d_in[9];
  const float* b_fc1 = (const float*)d_in[10];
  const float* W_fc2 = (const float*)d_in[11];
  const float* b_fc2 = (const float*)d_in[12];

  _Float16* h1 = (_Float16*)d_ws;  // (T,B,64) f16 inter-layer buffer, 64 MiB

  dim3 grid(kB / kBT);   // 64 workgroups
  dim3 block(kThreads);  // 4 waves

  lstm_layer<6, 1, true, false><<<grid, block, 0, stream>>>(
      x, W_ih0, W_hh0, b_ih0, b_hh0, h1,
      nullptr, nullptr, nullptr, nullptr, nullptr);

  lstm_layer<64, 2, false, true><<<grid, block, 0, stream>>>(
      h1, W_ih1, W_hh1, b_ih1, b_hh1, nullptr,
      W_fc1, b_fc1, W_fc2, b_fc2, (float*)d_out);
}
